// DioraBase_90022514524676
// MI455X (gfx1250) — matrix operand order repaired and checked by reading, not compiled, not run on hardware
//
#include <hip/hip_runtime.h>
#include <stdint.h>

typedef unsigned short ushort_t;
typedef __attribute__((ext_vector_type(16))) __bf16 v16bf;
typedef __attribute__((ext_vector_type(8)))  float  v8f;

#define B_     64
#define T_     24
#define D_     512
#define M_     36
#define NCELLS 300
#define EPSF   1e-8f

union UBF16 { v16bf v; ushort_t u[16]; };

__device__ __forceinline__ int off_of(int l) { return 24 * l - ((l * (l - 1)) >> 1); }

__device__ __forceinline__ ushort_t f2bf(float f) {
  uint32_t u = __builtin_bit_cast(uint32_t, f);
  uint32_t r = u + 0x7FFFu + ((u >> 16) & 1u);
  return (ushort_t)(r >> 16);
}
__device__ __forceinline__ float bf2f(ushort_t h) {
  uint32_t u = ((uint32_t)h) << 16;
  return __builtin_bit_cast(float, u);
}
__device__ __forceinline__ uint32_t pack2bf(float a, float b) {
  return (uint32_t)f2bf(a) | ((uint32_t)f2bf(b) << 16);
}

// ---------------------------------------------------------------------------
// WMMA GEMM: C(R x 512) = A(R x K, bf16, lda) @ Wt(512 x K, bf16, row = out col)
// optional bias (len 512), relu; out f32 or bf16.
// Block: 256 threads = 8 waves, tile 128x128; wave -> 32x64 subtile.
// Double-buffered LDS, b128 global loads / LDS stores.
// ---------------------------------------------------------------------------
__global__ __launch_bounds__(256) void wmma_gemm_n512(
    const ushort_t* __restrict__ A, int lda,
    const ushort_t* __restrict__ Wt,   // 512 x K, row-major (pre-transposed)
    const float* __restrict__ bias, int relu, int outBf16,
    void* __restrict__ outP, int R, int K)
{
  __shared__ ushort_t As[2][128][40];
  __shared__ ushort_t Bs[2][128][40];
  const int tid = threadIdx.x;
  const int rowBase = blockIdx.x * 128;
  const int colBase = blockIdx.y * 128;
  const int wave = tid >> 5, lane = tid & 31;
  const int wm = wave >> 1, wn = wave & 1;
  const int n16 = lane & 15, hh = lane >> 4;

  // tile loader assignment: each thread owns 16 contiguous bf16 of one row
  const int lr = tid >> 1;           // 0..127: tile row
  const int lc = (tid & 1) * 16;     // 0 or 16
  const int gr = rowBase + lr;       // global A row
  const uint4 z4 = {0u, 0u, 0u, 0u};
  uint4 a0 = z4, a1 = z4, b0 = z4, b1 = z4;

  auto loadTiles = [&](int k0) {
    if (gr < R) {
      const uint4* sA = (const uint4*)(A + (size_t)gr * lda + (k0 + lc));
      a0 = sA[0]; a1 = sA[1];
    } else {
      a0 = z4; a1 = z4;
    }
    const uint4* sB = (const uint4*)(Wt + (size_t)(colBase + lr) * K + (k0 + lc));
    b0 = sB[0]; b1 = sB[1];
  };
  auto storeTiles = [&](int buf) {
    *(uint4*)&As[buf][lr][lc]     = a0;
    *(uint4*)&As[buf][lr][lc + 8] = a1;
    *(uint4*)&Bs[buf][lr][lc]     = b0;
    *(uint4*)&Bs[buf][lr][lc + 8] = b1;
  };

  v8f acc[2][4];
#pragma unroll
  for (int mi = 0; mi < 2; ++mi)
#pragma unroll
    for (int ni = 0; ni < 4; ++ni)
#pragma unroll
      for (int q = 0; q < 8; ++q) acc[mi][ni][q] = 0.0f;

  loadTiles(0);
  storeTiles(0);
  int buf = 0;

  for (int k0 = 0; k0 < K; k0 += 32) {
    __syncthreads();
    const bool hasNext = (k0 + 32) < K;
    if (hasNext) loadTiles(k0 + 32);

    v16bf afrag[2], bfrag[4];
#pragma unroll
    for (int mi = 0; mi < 2; ++mi) {
      UBF16 ua;
      const int row = wm * 32 + mi * 16 + n16;
#pragma unroll
      for (int j = 0; j < 8; ++j) {
        const int kk = ((j >> 2) << 4) + (hh << 3) + ((j & 3) << 1);
        ua.u[2 * j]     = As[buf][row][kk];
        ua.u[2 * j + 1] = As[buf][row][kk + 1];
      }
      afrag[mi] = ua.v;
    }
#pragma unroll
    for (int ni = 0; ni < 4; ++ni) {
      UBF16 ub;
      const int col = wn * 64 + ni * 16 + n16;
#pragma unroll
      for (int j = 0; j < 8; ++j) {
        const int kk = ((j >> 2) << 4) + (hh << 3) + ((j & 3) << 1);
        ub.u[2 * j]     = Bs[buf][col][kk];
        ub.u[2 * j + 1] = Bs[buf][col][kk + 1];
      }
      bfrag[ni] = ub.v;
    }
#pragma unroll
    for (int mi = 0; mi < 2; ++mi)
#pragma unroll
      for (int ni = 0; ni < 4; ++ni)
        acc[mi][ni] = __builtin_amdgcn_wmma_f32_16x16x32_bf16(
            false, afrag[mi], false, bfrag[ni], (short)0, acc[mi][ni], false, false);

    if (hasNext) storeTiles(buf ^ 1);
    buf ^= 1;
  }

#pragma unroll
  for (int mi = 0; mi < 2; ++mi) {
#pragma unroll
    for (int ni = 0; ni < 4; ++ni) {
      const int col = colBase + wn * 64 + ni * 16 + n16;
      const float bv = bias ? bias[col] : 0.0f;
#pragma unroll
      for (int v = 0; v < 8; ++v) {
        const int row = rowBase + wm * 32 + mi * 16 + (hh << 3) + v;
        if (row < R) {
          float x = acc[mi][ni][v] + bv;
          if (relu) x = fmaxf(x, 0.0f);
          if (outBf16) ((ushort_t*)outP)[(size_t)row * 512 + col] = f2bf(x);
          else         ((float*)outP)[(size_t)row * 512 + col] = x;
        }
      }
    }
  }
}

// ---------------------------------------------------------------------------
// Gather kernels: build bf16 [lh | rh] rows (width 1024) + ls+rs scalar.
// Packed 2x bf16 per 32-bit store.
// ---------------------------------------------------------------------------
__global__ __launch_bounds__(256) void inside_gather_kernel(
    const float* __restrict__ inH, const float* __restrict__ inS,
    ushort_t* __restrict__ lhrh, float* __restrict__ lsrs,
    int level, int L, int N)
{
  const int r = blockIdx.x;
  const int n = r % N;
  const int i = (r / N) % L;
  const int b = r / (N * L);
  const int lcell = off_of(n) + i;
  const int rcell = off_of(level - 1 - n) + i + n + 1;
  const float* ls = inH + ((size_t)b * NCELLS + lcell) * 512;
  const float* rs = inH + ((size_t)b * NCELLS + rcell) * 512;
  uint32_t* dst = (uint32_t*)(lhrh + (size_t)r * 1024);
  const int tid = threadIdx.x;
  const int t2 = tid * 2;
  dst[tid]       = pack2bf(ls[t2], ls[t2 + 1]);
  dst[256 + tid] = pack2bf(rs[t2], rs[t2 + 1]);
  if (tid == 0)
    lsrs[r] = inS[(size_t)b * NCELLS + lcell] + inS[(size_t)b * NCELLS + rcell];
}

__global__ __launch_bounds__(256) void outside_gather_kernel(
    const float* __restrict__ inH, const float* __restrict__ inS,
    const float* __restrict__ outH, const float* __restrict__ outS,
    ushort_t* __restrict__ lhrh, float* __restrict__ lsrs,
    int level, int L, int N)
{
  const int r = blockIdx.x;
  const int n = r % N;
  const int i = (r / N) % L;
  const int b = r / (N * L);
  const int j = i + level;
  int pcell, scell;
  if (n < i) {               // span is right child; sibling span (a, i-1)
    const int a = n;
    pcell = off_of(j - a) + a;
    scell = off_of(i - 1 - a) + a;
  } else {                   // span is left child; sibling span (j+1, bb)
    const int bb = j + 1 + (n - i);
    pcell = off_of(bb - i) + i;
    scell = off_of(bb - j - 1) + j + 1;
  }
  // lh slot <- inside sibling, rh slot <- outside parent
  const float* ls = inH  + ((size_t)b * NCELLS + scell) * 512;
  const float* rs = outH + ((size_t)b * NCELLS + pcell) * 512;
  uint32_t* dst = (uint32_t*)(lhrh + (size_t)r * 1024);
  const int tid = threadIdx.x;
  const int t2 = tid * 2;
  dst[tid]       = pack2bf(ls[t2], ls[t2 + 1]);
  dst[256 + tid] = pack2bf(rs[t2], rs[t2 + 1]);
  if (tid == 0)
    lsrs[r] = inS[(size_t)b * NCELLS + scell] + outS[(size_t)b * NCELLS + pcell];
}

// ---------------------------------------------------------------------------
// score[r] = dot(t[r,:], rh_bf16[r,:]) + lsrs[r]   (one wave per row)
// ---------------------------------------------------------------------------
__global__ __launch_bounds__(256) void scoredot_kernel(
    const float* __restrict__ t, const ushort_t* __restrict__ lhrh,
    const float* __restrict__ lsrs, float* __restrict__ s, int R)
{
  const int wave = threadIdx.x >> 5, lane = threadIdx.x & 31;
  const int r = blockIdx.x * 8 + wave;
  if (r >= R) return;
  const float* tr = t + (size_t)r * 512;
  const ushort_t* rr = lhrh + (size_t)r * 1024 + 512;
  float acc = 0.f;
  for (int d = lane; d < 512; d += 32) acc += tr[d] * bf2f(rr[d]);
#pragma unroll
  for (int o = 16; o; o >>= 1) acc += __shfl_xor(acc, o, 32);
  if (lane == 0) s[r] = acc + lsrs[r];
}

// ---------------------------------------------------------------------------
// Per (b, span): softmax over N splits, weighted sum of h, unit-normalize.
// ---------------------------------------------------------------------------
__global__ __launch_bounds__(256) void agg_unit_kernel(
    const float* __restrict__ scores, const float* __restrict__ h,
    float* __restrict__ dest, int destStride, int destOff,
    float* __restrict__ sChart, int sOff, int L, int N)
{
  const int bi = blockIdx.x;
  const int b = bi / L, i = bi % L;
  const int r0 = bi * N;
  __shared__ float p[32];
  __shared__ float red[9];
  const int tid = threadIdx.x;
  if (tid == 0) {
    float mx = -1e30f;
    for (int n = 0; n < N; ++n) mx = fmaxf(mx, scores[r0 + n]);
    float ssum = 0.f;
    for (int n = 0; n < N; ++n) { float e = __expf(scores[r0 + n] - mx); p[n] = e; ssum += e; }
    const float inv = 1.f / ssum;
    float sa = 0.f;
    for (int n = 0; n < N; ++n) { p[n] *= inv; sa += p[n] * scores[r0 + n]; }
    sChart[(size_t)b * NCELLS + sOff + i] = sa;
  }
  __syncthreads();
  float o0 = 0.f, o1 = 0.f;
  for (int n = 0; n < N; ++n) {
    const float* hr = h + (size_t)(r0 + n) * 512;
    const float pn = p[n];
    o0 += pn * hr[tid];
    o1 += pn * hr[tid + 256];
  }
  float nr = o0 * o0 + o1 * o1;
  const int wave = tid >> 5, lane = tid & 31;
#pragma unroll
  for (int o = 16; o; o >>= 1) nr += __shfl_xor(nr, o, 32);
  if (lane == 0) red[wave] = nr;
  __syncthreads();
  if (tid == 0) { float t2 = 0.f; for (int w = 0; w < 8; ++w) t2 += red[w]; red[8] = 1.f / (sqrtf(t2) + EPSF); }
  __syncthreads();
  const float inv = red[8];
  float* dp = dest + ((size_t)b * destStride + destOff + i) * 512;
  dp[tid] = o0 * inv; dp[tid + 256] = o1 * inv;
}

// ---------------------------------------------------------------------------
// out = unit(hin + softmax(hin . obj) @ obj), written into chart cell.
// ---------------------------------------------------------------------------
__global__ __launch_bounds__(256) void atten_unit_kernel(
    const float* __restrict__ hin, const float* __restrict__ obj,
    float* __restrict__ chartH, int chartOff, int L)
{
  const int bi = blockIdx.x;
  const int b = bi / L, i = bi % L;
  __shared__ float hrow[512];
  __shared__ float sc[36];
  __shared__ float red[9];
  const int tid = threadIdx.x;
  hrow[tid]       = hin[(size_t)bi * 512 + tid];
  hrow[tid + 256] = hin[(size_t)bi * 512 + tid + 256];
  __syncthreads();
  const int wave = tid >> 5, lane = tid & 31;
  for (int m = wave; m < 36; m += 8) {
    const float* orow = obj + ((size_t)b * 36 + m) * 512;
    float acc = 0.f;
    for (int d = lane; d < 512; d += 32) acc += hrow[d] * orow[d];
#pragma unroll
    for (int o = 16; o; o >>= 1) acc += __shfl_xor(acc, o, 32);
    if (lane == 0) sc[m] = acc;
  }
  __syncthreads();
  if (tid == 0) {
    float mx = -1e30f;
    for (int m = 0; m < 36; ++m) mx = fmaxf(mx, sc[m]);
    float ssum = 0.f;
    for (int m = 0; m < 36; ++m) { float e = __expf(sc[m] - mx); sc[m] = e; ssum += e; }
    const float inv = 1.f / ssum;
    for (int m = 0; m < 36; ++m) sc[m] *= inv;
  }
  __syncthreads();
  float o0 = hrow[tid], o1 = hrow[tid + 256];
  for (int m = 0; m < 36; ++m) {
    const float* orow = obj + ((size_t)b * 36 + m) * 512;
    const float pm = sc[m];
    o0 += pm * orow[tid];
    o1 += pm * orow[tid + 256];
  }
  float nr = o0 * o0 + o1 * o1;
#pragma unroll
  for (int o = 16; o; o >>= 1) nr += __shfl_xor(nr, o, 32);
  if (lane == 0) red[wave] = nr;
  __syncthreads();
  if (tid == 0) { float t2 = 0.f; for (int w = 0; w < 8; ++w) t2 += red[w]; red[8] = 1.f / (sqrtf(t2) + EPSF); }
  __syncthreads();
  const float inv = red[8];
  float* dp = chartH + ((size_t)b * NCELLS + chartOff + i) * 512;
  dp[tid] = o0 * inv; dp[tid + 256] = o1 * inv;
}

__global__ __launch_bounds__(256) void unit_inplace_kernel(float* __restrict__ rows)
{
  float* pr = rows + (size_t)blockIdx.x * 512;
  const int tid = threadIdx.x;
  float v0 = pr[tid], v1 = pr[tid + 256];
  float nr = v0 * v0 + v1 * v1;
  const int wave = tid >> 5, lane = tid & 31;
  __shared__ float red[9];
#pragma unroll
  for (int o = 16; o; o >>= 1) nr += __shfl_xor(nr, o, 32);
  if (lane == 0) red[wave] = nr;
  __syncthreads();
  if (tid == 0) { float t2 = 0.f; for (int w = 0; w < 8; ++w) t2 += red[w]; red[8] = 1.f / (sqrtf(t2) + EPSF); }
  __syncthreads();
  const float inv = red[8];
  pr[tid] = v0 * inv; pr[tid + 256] = v1 * inv;
}

__global__ __launch_bounds__(256) void root_kernel(
    const float* __restrict__ root, float* __restrict__ outH)
{
  const int tid = threadIdx.x;
  float v0 = root[tid], v1 = root[tid + 256];
  float nr = v0 * v0 + v1 * v1;
  const int wave = tid >> 5, lane = tid & 31;
  __shared__ float red[9];
#pragma unroll
  for (int o = 16; o; o >>= 1) nr += __shfl_xor(nr, o, 32);
  if (lane == 0) red[wave] = nr;
  __syncthreads();
  if (tid == 0) { float t2 = 0.f; for (int w = 0; w < 8; ++w) t2 += red[w]; red[8] = 1.f / (sqrtf(t2) + EPSF); }
  __syncthreads();
  const float inv = red[8];
  float* dp = outH + ((size_t)blockIdx.x * NCELLS + (NCELLS - 1)) * 512;
  dp[tid] = v0 * inv; dp[tid + 256] = v1 * inv;
}

__global__ void f32_to_bf16_kernel(const float* __restrict__ src, ushort_t* __restrict__ dst, int n)
{
  const int idx = blockIdx.x * 256 + threadIdx.x;
  if (idx < n) dst[idx] = f2bf(src[idx]);
}

// dst (512 x K) = transpose of src (K x 512), bf16
__global__ void transpose_bf16_kernel(const float* __restrict__ src, ushort_t* __restrict__ dst, int K)
{
  const int idx = blockIdx.x * 256 + threadIdx.x;
  if (idx < K * 512) {
    const int k = idx >> 9, n = idx & 511;
    dst[(size_t)n * K + k] = f2bf(src[idx]);
  }
}

// wcat_t (512 x 1024): row n holds [W0l[:,n] ; W0r[:,n]]
__global__ void build_wcat_t_kernel(const float* __restrict__ w0l, const float* __restrict__ w0r,
                                    ushort_t* __restrict__ dst)
{
  const int idx = blockIdx.x * 256 + threadIdx.x;
  if (idx < 1024 * 512) {
    const int k = idx >> 9, n = idx & 511;
    const float v = (k < 512) ? w0l[(size_t)k * 512 + n] : w0r[(size_t)(k - 512) * 512 + n];
    dst[(size_t)n * 1024 + k] = f2bf(v);
  }
}

__global__ void zero_f32_kernel(float* __restrict__ p, int n)
{
  const int idx = blockIdx.x * 256 + threadIdx.x;
  if (idx < n) p[idx] = 0.f;
}

// ---------------------------------------------------------------------------
extern "C" void kernel_launch(void* const* d_in, const int* in_sizes, int n_in,
                              void* d_out, int out_size, void* d_ws, size_t ws_size,
                              hipStream_t stream)
{
  (void)in_sizes; (void)n_in; (void)out_size; (void)ws_size;
  const float* x      = (const float*)d_in[0];
  const float* obj    = (const float*)d_in[1];
  const float* W_leaf = (const float*)d_in[2];
  const float* b_leaf = (const float*)d_in[3];
  const float* W0l    = (const float*)d_in[4];
  const float* W0r    = (const float*)d_in[5];
  const float* B0v    = (const float*)d_in[6];
  const float* W1     = (const float*)d_in[7];
  const float* B1v    = (const float*)d_in[8];
  const float* Smat   = (const float*)d_in[9];
  const float* root_h = (const float*)d_in[10];

  float* insideH  = (float*)d_out;                           // (B, 300, 512)
  float* outsideH = insideH + (size_t)B_ * NCELLS * D_;      // (B, 300, 512)

  char* wp = (char*)d_ws;
  auto carve = [&](size_t bytes) -> void* {
    void* q = (void*)wp;
    wp += (bytes + 255) & ~(size_t)255;
    return q;
  };
  const int Rmax = B_ * 24 * 23;                              // 35328
  float*    insideS  = (float*)carve((size_t)B_ * NCELLS * 4);
  float*    outsideS = (float*)carve((size_t)B_ * NCELLS * 4);
  ushort_t* lhrh     = (ushort_t*)carve((size_t)Rmax * 1024 * 2);
  ushort_t* hmid     = (ushort_t*)carve((size_t)Rmax * 512 * 2);
  float*    tbuf     = (float*)carve((size_t)Rmax * 512 * 4); // score t, then compose2 out
  float*    scores   = (float*)carve((size_t)Rmax * 4);
  float*    lsrs     = (float*)carve((size_t)Rmax * 4);
  float*    hagg     = (float*)carve((size_t)B_ * T_ * D_ * 4);
  float*    leafbuf  = (float*)carve((size_t)B_ * T_ * D_ * 4);
  ushort_t* xbf      = (ushort_t*)carve((size_t)B_ * T_ * D_ * 2);
  ushort_t* wcat_t   = (ushort_t*)carve((size_t)1024 * 512 * 2); // 512 x 1024
  ushort_t* w1t      = (ushort_t*)carve((size_t)512 * 512 * 2);  // 512 x 512 (T)
  ushort_t* st       = (ushort_t*)carve((size_t)512 * 512 * 2);  // 512 x 512 (T)
  ushort_t* wlt      = (ushort_t*)carve((size_t)512 * 512 * 2);  // 512 x 512 (T)

  auto gemm = [&](const ushort_t* A, int lda, const ushort_t* Wt, const float* bias,
                  int relu, int outBf16, void* out, int R, int K) {
    dim3 g((unsigned)((R + 127) / 128), 4);
    wmma_gemm_n512<<<g, dim3(256), 0, stream>>>(A, lda, Wt, bias, relu, outBf16, out, R, K);
  };

  // --- init + weight conversion ---
  zero_f32_kernel<<<(B_ * NCELLS + 255) / 256, 256, 0, stream>>>(insideS, B_ * NCELLS);
  zero_f32_kernel<<<(B_ * NCELLS + 255) / 256, 256, 0, stream>>>(outsideS, B_ * NCELLS);
  f32_to_bf16_kernel<<<(B_ * T_ * D_ + 255) / 256, 256, 0, stream>>>(x, xbf, B_ * T_ * D_);
  transpose_bf16_kernel<<<(512 * 512 + 255) / 256, 256, 0, stream>>>(W_leaf, wlt, 512);
  transpose_bf16_kernel<<<(512 * 512 + 255) / 256, 256, 0, stream>>>(W1, w1t, 512);
  transpose_bf16_kernel<<<(512 * 512 + 255) / 256, 256, 0, stream>>>(Smat, st, 512);
  build_wcat_t_kernel<<<(1024 * 512 + 255) / 256, 256, 0, stream>>>(W0l, W0r, wcat_t);

  // --- leaf level (level 0) ---
  gemm(xbf, 512, wlt, b_leaf, 1, 0, leafbuf, B_ * T_, 512);
  unit_inplace_kernel<<<B_ * T_, 256, 0, stream>>>(leafbuf);
  atten_unit_kernel<<<B_ * T_, 256, 0, stream>>>(leafbuf, obj, insideH, 0, T_);

  // --- inside pass ---
  for (int level = 1; level < T_; ++level) {
    const int L = T_ - level, N = level;
    const int R = B_ * L * N;
    const int offl = 24 * level - level * (level - 1) / 2;
    inside_gather_kernel<<<R, 256, 0, stream>>>(insideH, insideS, lhrh, lsrs, level, L, N);
    gemm(lhrh, 1024, st, nullptr, 0, 0, tbuf, R, 512);                 // t = lh @ S
    scoredot_kernel<<<(R + 7) / 8, 256, 0, stream>>>(tbuf, lhrh, lsrs, scores, R);
    gemm(lhrh, 1024, wcat_t, B0v, 1, 1, hmid, R, 1024);                // relu(lh@W0l+rh@W0r+B0)
    gemm(hmid, 512, w1t, B1v, 1, 0, tbuf, R, 512);                     // relu(hmid@W1+B1)
    agg_unit_kernel<<<B_ * L, 256, 0, stream>>>(scores, tbuf, hagg, L, 0, insideS, offl, L, N);
    atten_unit_kernel<<<B_ * L, 256, 0, stream>>>(hagg, obj, insideH, offl, L);
  }

  // --- outside init (root) ---
  root_kernel<<<B_, 256, 0, stream>>>(root_h, outsideH);

  // --- outside pass ---
  for (int level = T_ - 2; level >= 0; --level) {
    const int L = T_ - level, N = T_ - level - 1;
    const int R = B_ * L * N;
    const int offl = 24 * level - level * (level - 1) / 2;
    outside_gather_kernel<<<R, 256, 0, stream>>>(insideH, insideS, outsideH, outsideS,
                                                 lhrh, lsrs, level, L, N);
    gemm(lhrh, 1024, st, nullptr, 0, 0, tbuf, R, 512);                 // t = sh @ S
    scoredot_kernel<<<(R + 7) / 8, 256, 0, stream>>>(tbuf, lhrh, lsrs, scores, R);
    gemm(lhrh, 1024, wcat_t, B0v, 1, 1, hmid, R, 1024);                // compose(sh, ph)
    gemm(hmid, 512, w1t, B1v, 1, 0, tbuf, R, 512);
    agg_unit_kernel<<<B_ * L, 256, 0, stream>>>(scores, tbuf, outsideH, NCELLS, offl,
                                                outsideS, offl, L, N);
  }
}
